// CustomDotProductAttention_1563368095789
// MI455X (gfx1250) — compile-verified, matrix operand order checked by reading, not run on hardware
//
#include <hip/hip_runtime.h>

// Fused causal flash-attention forward for MI455X (gfx1250, wave32).
// L=S=2048, B=2, H=16, D=128, fp32 in/out; bf16 WMMA matmuls, fp32 softmax.
// Double-buffered LDS K/V tiles; row-sums via WMMA against a ones matrix;
// exp2-domain online softmax (log2e folded into Q).

#define L_DIM 2048
#define BH    32            // B*H
#define D_DIM 128
#define BM    128           // query rows per workgroup (8 waves x 16 rows)
#define BN    32            // keys per S-tile
#define NW    8             // waves per workgroup

typedef __attribute__((ext_vector_type(16))) __bf16 v16bf;
typedef __attribute__((ext_vector_type(8)))  float  v8f;

union Frag16 { v16bf v; unsigned u[8]; };

__device__ __forceinline__ unsigned short bf16_bits(float x) {
  __bf16 h = (__bf16)x;                 // backend picks v_cvt_*_bf16_f32
  return __builtin_bit_cast(unsigned short, h);
}
__device__ __forceinline__ unsigned pack2_bf16(float a, float b) {
  return (unsigned)bf16_bits(a) | ((unsigned)bf16_bits(b) << 16);
}
// 16-bit A/B fragment pair base (K index) for vgpr v, lane-half hf.
// A 16x32 layout: lanes 0-15: v0..3 -> K 0..7, v4..7 -> K 16..23;
//                 lanes 16-31: v0..3 -> K 8..15, v4..7 -> K 24..31.
__device__ __forceinline__ int pair_k0(int v, int hf) {
  return (v < 4) ? (2 * v + 8 * hf) : (16 + 2 * (v - 4) + 8 * hf);
}

__global__ __launch_bounds__(256, 1)
void fa_fwd_causal(const float* __restrict__ Q, const float* __restrict__ K,
                   const float* __restrict__ V, float* __restrict__ O) {
  // double-buffered K tile: bf16 pairs packed along d (QK^T contraction dim)
  __shared__ unsigned sK[2][BN][D_DIM / 2 + 1];
  // double-buffered V tile transposed: [d][key], PV contraction pairs contiguous
  __shared__ alignas(4) unsigned short sVt[2][D_DIM][BN + 2];
  // per-wave P transpose buffer (C-layout -> A-layout)
  __shared__ alignas(4) unsigned short sP[NW][16][BN + 2];

  const int tid  = threadIdx.x;
  const int wave = tid >> 5;
  const int lane = tid & 31;
  const int nl   = lane & 15;
  const int hf   = lane >> 4;

  const int bh  = blockIdx.y;            // fused batch*head index
  const int wgr = blockIdx.x * BM;       // first query row of workgroup
  const int wr  = wgr + wave * 16;       // first query row of this wave

  const size_t rstride = (size_t)BH * D_DIM;   // floats between tokens

  // ---- load Q fragments (A-layout); fold log2(e)/sqrt(D) in ----
  const float qscale2 = 0.12751743f;     // (1/sqrt(128)) * log2(e)
  Frag16 aq[4];
  {
    const float* qp = Q + (size_t)(wr + nl) * rstride + (size_t)bh * D_DIM;
#pragma unroll
    for (int c = 0; c < 4; ++c) {
#pragma unroll
      for (int v = 0; v < 8; ++v) {
        const int k0 = pair_k0(v, hf);
        const float2 f = *reinterpret_cast<const float2*>(qp + 32 * c + k0);
        aq[c].u[v] = pack2_bf16(f.x * qscale2, f.y * qscale2);
      }
    }
  }

  // ones B-matrix (32x16 of bf16 1.0) for WMMA row-sums
  Frag16 ones;
#pragma unroll
  for (int v = 0; v < 8; ++v) ones.u[v] = 0x3F803F80u;

  v8f o[8];
#pragma unroll
  for (int f = 0; f < 8; ++f) o[f] = {};
  float mrow[8], lrow[8];
#pragma unroll
  for (int v = 0; v < 8; ++v) { mrow[v] = -1e30f; lrow[v] = 0.0f; }

  // ---- cooperative loader: 256 threads x 16 floats of each of K and V ----
  const int kr = tid >> 3;               // key row 0..31
  const int dc = (tid & 7) * 16;         // d column base
  const float* kbase = K + (size_t)bh * D_DIM + (size_t)kr * rstride + dc;
  const float* vbase = V + (size_t)bh * D_DIM + (size_t)kr * rstride + dc;

  float4 kq[4], vq[4];
  auto issue = [&](int s0) {             // global loads only (stay in flight)
    const float4* kp = reinterpret_cast<const float4*>(kbase + (size_t)s0 * rstride);
    const float4* vp = reinterpret_cast<const float4*>(vbase + (size_t)s0 * rstride);
#pragma unroll
    for (int q = 0; q < 4; ++q) { kq[q] = kp[q]; vq[q] = vp[q]; }
  };
  auto commit = [&](int buf) {           // convert + LDS store
    float kf[16], vf[16];
#pragma unroll
    for (int q = 0; q < 4; ++q) {
      kf[4*q+0]=kq[q].x; kf[4*q+1]=kq[q].y; kf[4*q+2]=kq[q].z; kf[4*q+3]=kq[q].w;
      vf[4*q+0]=vq[q].x; vf[4*q+1]=vq[q].y; vf[4*q+2]=vq[q].z; vf[4*q+3]=vq[q].w;
    }
#pragma unroll
    for (int j = 0; j < 8; ++j) sK[buf][kr][dc / 2 + j] = pack2_bf16(kf[2*j], kf[2*j+1]);
#pragma unroll
    for (int j = 0; j < 16; ++j) sVt[buf][dc + j][kr] = bf16_bits(vf[j]);
  };

  const int ntiles = (wgr + BM) / BN;    // causal: keys up to last row of WG
  issue(0);
  commit(0);

  for (int t = 0; t < ntiles; ++t) {
    __syncthreads();                     // buffer t&1 visible to all waves
    const int s0  = t * BN;
    const int cur = t & 1;
    const bool havenext = (t + 1 < ntiles);
    if (havenext) issue(s0 + BN);        // overlap HBM latency with compute
    if (t + 2 < ntiles)
      __builtin_prefetch(kbase + (size_t)(s0 + 2 * BN) * rstride, 0, 1);

    if (s0 <= wr + 15) {                 // tile not entirely in the future
      // ---- scores in log2 domain: S = (Q*scale*log2e) @ K^T, 16x32 ----
      v8f sc[2];
#pragma unroll
      for (int t2 = 0; t2 < 2; ++t2) {
        v8f acc = {};
#pragma unroll
        for (int c = 0; c < 4; ++c) {
          Frag16 bk;
#pragma unroll
          for (int v = 0; v < 8; ++v)
            bk.u[v] = sK[cur][t2 * 16 + nl][16 * c + (pair_k0(v, hf) >> 1)];
          acc = __builtin_amdgcn_wmma_f32_16x16x32_bf16(false, aq[c].v, false, bk.v,
                                                        (short)0, acc, false, false);
        }
        sc[t2] = acc;
      }

      // ---- causal mask + online softmax (max over 16 lanes per row) ----
      const bool diag = (s0 + BN > wr);
      float corr[8];
#pragma unroll
      for (int v = 0; v < 8; ++v) {
        const int i = wr + v + 8 * hf;   // global query row of this C slot
        float s0v = sc[0][v], s1v = sc[1][v];
        if (diag) {
          if (s0 + nl      > i) s0v = -1e30f;
          if (s0 + 16 + nl > i) s1v = -1e30f;
        }
        float rm = fmaxf(s0v, s1v);
#pragma unroll
        for (int off = 1; off < 16; off <<= 1)
          rm = fmaxf(rm, __shfl_xor(rm, off, 16));
        const float mn = fmaxf(mrow[v], rm);
        corr[v] = exp2f(mrow[v] - mn);   // single v_exp_f32
        mrow[v] = mn;
        sc[0][v] = exp2f(s0v - mn);
        sc[1][v] = exp2f(s1v - mn);
#pragma unroll
        for (int f = 0; f < 8; ++f) o[f][v] *= corr[v];
      }

      // ---- transpose P: C-layout -> A-layout via per-wave LDS ----
#pragma unroll
      for (int v = 0; v < 8; ++v) {
        sP[wave][v + 8 * hf][nl]      = bf16_bits(sc[0][v]);
        sP[wave][v + 8 * hf][16 + nl] = bf16_bits(sc[1][v]);
      }
      __builtin_amdgcn_wave_barrier();
      asm volatile("s_wait_dscnt 0" ::: "memory");  // DS in-order per wave
      __builtin_amdgcn_wave_barrier();
      Frag16 ap;
#pragma unroll
      for (int v = 0; v < 8; ++v) {
        const unsigned* prow = reinterpret_cast<const unsigned*>(&sP[wave][nl][0]);
        ap.u[v] = prow[pair_k0(v, hf) >> 1];
      }

      // ---- row sums of (bf16-rounded) P via WMMA against ones ----
      {
        v8f rsf = {};
        rsf = __builtin_amdgcn_wmma_f32_16x16x32_bf16(false, ap.v, false, ones.v,
                                                      (short)0, rsf, false, false);
#pragma unroll
        for (int v = 0; v < 8; ++v)
          lrow[v] = lrow[v] * corr[v] + rsf[v];
      }

      // ---- O += P @ V over 8 d-subtiles ----
#pragma unroll
      for (int nt = 0; nt < 8; ++nt) {
        Frag16 bv;
#pragma unroll
        for (int v = 0; v < 8; ++v) {
          const unsigned* vrow =
              reinterpret_cast<const unsigned*>(&sVt[cur][nt * 16 + nl][0]);
          bv.u[v] = vrow[pair_k0(v, hf) >> 1];
        }
        o[nt] = __builtin_amdgcn_wmma_f32_16x16x32_bf16(false, ap.v, false, bv.v,
                                                        (short)0, o[nt], false, false);
      }
    }

    if (havenext) commit(cur ^ 1);       // stage next tile into the other buffer
  }

  // ---- epilogue: normalize and store ----
#pragma unroll
  for (int v = 0; v < 8; ++v) {
    const float inv = 1.0f / lrow[v];
    const int i = wr + v + 8 * hf;
    float* op = O + (size_t)i * rstride + (size_t)bh * D_DIM;
#pragma unroll
    for (int f = 0; f < 8; ++f)
      op[f * 16 + nl] = o[f][v] * inv;
  }
}

extern "C" void kernel_launch(void* const* d_in, const int* in_sizes, int n_in,
                              void* d_out, int out_size, void* d_ws, size_t ws_size,
                              hipStream_t stream) {
  (void)in_sizes; (void)n_in; (void)d_ws; (void)ws_size; (void)out_size;
  const float* Q = (const float*)d_in[0];
  const float* K = (const float*)d_in[1];
  const float* V = (const float*)d_in[2];
  float* Out = (float*)d_out;
  dim3 grid(L_DIM / BM, BH);
  fa_fwd_causal<<<grid, dim3(256), 0, stream>>>(Q, K, V, Out);
}
